// RNN_19018115186910
// MI455X (gfx1250) — matrix operand order repaired and checked by reading, not compiled, not run on hardware
//
#include <hip/hip_runtime.h>
#include <hip/hip_bf16.h>
#include <math.h>

// ---------------------------------------------------------------------------
// RNN scan, MI455X (gfx1250, wave32) persistent-kernel design.
//   INPUT_SIZE=256, HIDDEN=1024, N_LABELS=32, T=16384
// 64 persistent workgroups x 128 threads (4 waves). WG b owns output rows
// [16b, 16b+16). Fused weight panel W = [Wi | Wh] (16 x 1280) is converted to
// bf16 and pre-packed into LDS in v_wmma_f32_16x16x32_bf16 A-fragment order.
// Per step: stage vecB = [x_t ; h_prev] (bf16) in LDS, run 40 K-steps of WMMA
// split 10/10/10/10 across the four waves (shorter dependent XDL chain), LDS
// reduce, tanh, publish 16 h values, grid-wide release/acquire barrier.
//
// NOTE: B is deliberately NOT masked to column N=0. Every lane loads the same
// 16 bf16 K-values, so all 16 columns of D compute identical copies of the
// matvec; we extract column 0 (lanes 0/16). This keeps the serial chain to
// ds_load -> v_wmma with zero VALU masking overhead.
// ---------------------------------------------------------------------------

#define IN_SZ   256
#define HID     1024
#define NLAB    32
#define TSTEPS  16384
#define KTOT    (IN_SZ + HID)     // 1280
#define KSTEPS  (KTOT / 32)       // 40
#define NWAVES  4
#define KPART   (KSTEPS / NWAVES) // 10 k-steps per wave
#define NWG     64                // 64 WGs x 16 rows = 1024 rows
#define BLOCK   (NWAVES * 32)     // 128 threads

typedef __bf16  v16bf __attribute__((ext_vector_type(16)));
typedef __bf16  v4bf  __attribute__((ext_vector_type(4)));
typedef float   v8f   __attribute__((ext_vector_type(8)));
typedef float   v4f   __attribute__((ext_vector_type(4)));

// ---------------------------------------------------------------- init ------
__global__ void rnn_init_kernel(float* hbuf, int* bar) {
    int tid = blockIdx.x * blockDim.x + threadIdx.x;
    for (int i = tid; i < 2 * HID; i += gridDim.x * blockDim.x) hbuf[i] = 0.0f;
    if (tid == 0) *bar = 0;
}

// ---------------------------------------------------------------- scan ------
__global__ void __launch_bounds__(BLOCK)
rnn_scan_kernel(const float* __restrict__ line,
                const float* __restrict__ Wi,
                const float* __restrict__ bi,
                const float* __restrict__ Wh,
                const float* __restrict__ bh,
                float* __restrict__ hbuf,   // [2][HID] double buffer
                int*   __restrict__ bar) {
    // A fragments: [kstep][lane] -> 16 bf16 (32B), total 40*32*32 = 40 KB
    __shared__ v16bf Apack[KSTEPS][32];
    __shared__ __align__(32) __bf16 vecB[KTOT];     // staged [x_t ; h_prev]
    __shared__ float partial[NWAVES - 1][16];       // split-K reduce -> wave0

    const int tid  = threadIdx.x;
    const int lane = tid & 31;
    const int wave = tid >> 5;
    const int rowbase = blockIdx.x * 16;

    // ---- pre-pack the fused bf16 weight panel into WMMA A-fragment order ----
    // 16-bit A 16x32 layout: m = lane&15 ; contiguous k order
    //   elems 0..15 -> K = 32s + (lane>>4)*16 + j
    {
        const int m  = rowbase + (lane & 15);
        const int kb = (lane >> 4) * 8;
        for (int s = wave * KPART; s < (wave + 1) * KPART; ++s) {
            v16bf frag;
#pragma unroll
            for (int j = 0; j < 16; ++j) {
                int k = s * 32 + (lane >> 4) * 16 + j;
                float v = (k < IN_SZ) ? Wi[m * IN_SZ + k]
                                      : Wh[m * HID + (k - IN_SZ)];
                frag[j] = (__bf16)v;
            }
            Apack[s][lane] = frag;
        }
    }

    // fused bias bi+bh for this lane's 8 output rows (used by lanes 0/16 of
    // wave0 only; computed everywhere, divergence-free)
    float biasv[8];
    {
        const int mb = (lane >> 4) * 8;
#pragma unroll
        for (int r = 0; r < 8; ++r) {
            int row = rowbase + mb + r;
            biasv[r] = bi[row] + bh[row];
        }
    }

    __syncthreads();

    for (int t = 0; t < TSTEPS; ++t) {
        const float* hprev = hbuf + (t & 1) * HID;
        float*       hnext = hbuf + ((t + 1) & 1) * HID;

        // ---- stage vecB = bf16([x_t ; h_prev]): float4 loads, b64 LDS stores
        for (int g = tid; g < KTOT / 4; g += BLOCK) {
            const int i = g * 4;                    // IN_SZ is a multiple of 4
            v4f v = (i < IN_SZ) ? *(const v4f*)&line[t * IN_SZ + i]
                                : *(const v4f*)&hprev[i - IN_SZ];
            v4bf b;
#pragma unroll
            for (int j = 0; j < 4; ++j) b[j] = (__bf16)v[j];
            *(v4bf*)&vecB[i] = b;
        }
        if (t + 1 < TSTEPS && tid < 64)             // next x row toward L2/L0
            __builtin_prefetch(&line[(t + 1) * IN_SZ] + (tid << 2), 0, 1);
        __syncthreads();

        // ---- split-K WMMA: this wave covers k-steps [wave*10, wave*10+10) --
        // B fragment: all lanes load the same 16 K-values (LDS broadcast per
        // half-wave) -> every D column equals the matvec; no masking needed.
        v8f c;
#pragma unroll
        for (int r = 0; r < 8; ++r) c[r] = 0.0f;

        for (int s = wave * KPART; s < (wave + 1) * KPART; ++s) {
            v16bf a = Apack[s][lane];
            v16bf b = *(const v16bf*)&vecB[s * 32 + ((lane >> 4) << 4)];
            c = __builtin_amdgcn_wmma_f32_16x16x32_bf16(
                    false, a, false, b, (short)0, c, false, false);
        }

        // ---- reduce the four K-parts; D column 0 = lanes 0 (M0..7)/16 (M8..15)
        const int mb = (lane >> 4) * 8;
        if (wave != 0 && (lane & 15) == 0) {
#pragma unroll
            for (int r = 0; r < 8; ++r) partial[wave - 1][mb + r] = c[r];
        }
        __syncthreads();
        if (wave == 0 && (lane & 15) == 0) {
#pragma unroll
            for (int r = 0; r < 8; ++r) {
                float s = c[r] + partial[0][mb + r] + partial[1][mb + r]
                               + partial[2][mb + r] + biasv[r];
                hnext[rowbase + mb + r] = tanhf(s);
            }
        }

        // ---- grid-wide release/acquire barrier (monotonic counter) --------
        if (tid == 0) {
            __threadfence();
            __hip_atomic_fetch_add(bar, 1, __ATOMIC_RELEASE,
                                   __HIP_MEMORY_SCOPE_AGENT);
            const int target = NWG * (t + 1);
            while (__hip_atomic_load(bar, __ATOMIC_ACQUIRE,
                                     __HIP_MEMORY_SCOPE_AGENT) < target)
                __builtin_amdgcn_s_sleep(2);
        }
        __syncthreads();   // also orders LDS reuse (vecB/partial) across steps
    }
    // final h lives in hbuf[0] (T even -> last write hits buffer 0)
}

// ---------------------------------------------------------------- output ----
__global__ void rnn_out_kernel(const float* __restrict__ h,     // hbuf[0]
                               const float* __restrict__ Wo,
                               const float* __restrict__ bo,
                               float* __restrict__ out) {
    __shared__ float logits[NLAB];
    __shared__ float logz;
    const int tid = threadIdx.x, lane = tid & 31, wave = tid >> 5;  // 8 waves

    for (int o = wave; o < NLAB; o += 8) {
        float s = 0.0f;
        for (int k = lane; k < HID; k += 32) s += h[k] * Wo[o * HID + k];
        for (int off = 16; off > 0; off >>= 1) s += __shfl_down(s, off, 32);
        if (lane == 0) logits[o] = s + bo[o];
    }
    __syncthreads();
    if (tid == 0) {
        float mx = logits[0];
        for (int i = 1; i < NLAB; ++i) mx = fmaxf(mx, logits[i]);
        float se = 0.0f;
        for (int i = 0; i < NLAB; ++i) se += expf(logits[i] - mx);
        logz = mx + logf(se);
    }
    __syncthreads();
    if (tid < NLAB) out[tid] = logits[tid] - logz;
}

// ---------------------------------------------------------------- launch ----
extern "C" void kernel_launch(void* const* d_in, const int* in_sizes, int n_in,
                              void* d_out, int out_size, void* d_ws, size_t ws_size,
                              hipStream_t stream) {
    const float* line = (const float*)d_in[0];  // [T,1,256]
    const float* Wi   = (const float*)d_in[1];  // [1024,256]
    const float* bi   = (const float*)d_in[2];  // [1024]
    const float* Wh   = (const float*)d_in[3];  // [1024,1024]
    const float* bh   = (const float*)d_in[4];  // [1024]
    const float* Wo   = (const float*)d_in[5];  // [32,1024]
    const float* bo   = (const float*)d_in[6];  // [32]

    float* hbuf = (float*)d_ws;                       // 2*1024 f32 = 8 KB
    int*   bar  = (int*)((char*)d_ws + 2 * HID * sizeof(float));

    rnn_init_kernel<<<8, 256, 0, stream>>>(hbuf, bar);
    rnn_scan_kernel<<<NWG, BLOCK, 0, stream>>>(line, Wi, bi, Wh, bh, hbuf, bar);
    rnn_out_kernel<<<1, 256, 0, stream>>>(hbuf, Wo, bo, (float*)d_out);
}
